// DocREModel_63617055588448
// MI455X (gfx1250) — matrix operand order repaired and checked by reading, not compiled, not run on hardware
//
#include <hip/hip_runtime.h>
#include <hip/hip_bf16.h>
#include <math.h>

typedef __attribute__((ext_vector_type(16))) _Float16 v16h;
typedef __attribute__((ext_vector_type(8)))  _Float16 v8h;
typedef __attribute__((ext_vector_type(8)))  float    v8f;
typedef __attribute__((ext_vector_type(4)))  float    v4f;
typedef __attribute__((ext_vector_type(4)))  int      v4i;

#define B_      4
#define L_      1024
#define D_      768
#define H_      12
#define NE_     42
#define M_      4
#define P_      512
#define BLK_    64
#define NL_     97
#define NPAD_   112            // NL_ padded to 7 WMMA n-tiles
#define OFFSET_ 1
#define KEX_    (2 * D_)       // 1536
#define KBI_    (D_ * BLK_)    // 49152

#define AS1 __attribute__((address_space(1)))
#define AS3 __attribute__((address_space(3)))

#if __has_builtin(__builtin_amdgcn_global_load_async_to_lds_b128) && \
    __has_builtin(__builtin_amdgcn_s_wait_asynccnt)
#define USE_ASYNC_LDS 1
#endif

// ---- fragment loaders: operands pre-packed f16, contiguous in K ------------
// A (16x32 f16): lane elements 0..7  <-> k = hi*8 .. hi*8+7
//                lane elements 8..15 <-> k = 16+hi*8 .. 16+hi*8+7
// B (32x16 f16): same k order as A (hiB = hi*16, so hiB/2 = hi*8)
__device__ __forceinline__ v16h frag_cat(v8h lo, v8h hi) {
    union { v16h v; v8h h[2]; } u;
    u.h[0] = lo; u.h[1] = hi;
    return u.v;
}
__device__ __forceinline__ v16h load_fragA(const _Float16* p, int hiA) {
    return frag_cat(*(const v8h*)(p + hiA), *(const v8h*)(p + 16 + hiA));
}
__device__ __forceinline__ v16h load_fragB(const _Float16* p, int hiB) {
    return frag_cat(*(const v8h*)(p + hiB / 2), *(const v8h*)(p + 16 + hiB / 2));
}

// ---------------------------------------------------------------------------
// Pack B-panels: out[n][k] = (n < Nsrc) ? in[k][n] : 0, f32 -> f16.
// Run once per weight; reads are strided but this is a one-shot pass.
// ---------------------------------------------------------------------------
__global__ void weight_transpose_kernel(const float* __restrict__ W,
                                        _Float16* __restrict__ Wt,
                                        int K, int Nsrc) {
    int n = blockIdx.y;
    _Float16* out = Wt + (size_t)n * K;
    if (n < Nsrc) {
        for (int k = blockIdx.x * blockDim.x + threadIdx.x; k < K;
             k += gridDim.x * blockDim.x)
            out[k] = (_Float16)W[(size_t)k * Nsrc + n];
    } else {
        for (int k = blockIdx.x * blockDim.x + threadIdx.x; k < K;
             k += gridDim.x * blockDim.x)
            out[k] = (_Float16)0.0f;
    }
}

// ---------------------------------------------------------------------------
// seqT[b][d][l] = (f16) seq[b][l][d]  — 32x32 tiled LDS transpose; the LDS
// fill uses CDNA5 async global->LDS copies when the toolchain exposes them.
// ---------------------------------------------------------------------------
__global__ void seq_transpose_kernel(const float* __restrict__ seq,
                                     _Float16* __restrict__ seqT) {
    __shared__ float tile[32][36];  // 36-float stride: 16B-aligned rows
    int lt = blockIdx.x, dt = blockIdx.y, b = blockIdx.z;
    int t = threadIdx.x;
    int r = t >> 3, q = t & 7;      // 32 rows x 8 float4 each
    const float* gsrc =
        seq + ((size_t)b * L_ + lt * 32 + r) * D_ + dt * 32 + q * 4;
#if defined(USE_ASYNC_LDS)
    __builtin_amdgcn_global_load_async_to_lds_b128(
        (AS1 v4i*)(unsigned long long)(const void*)gsrc,
        (AS3 v4i*)(unsigned)(unsigned long long)(void*)&tile[r][q * 4], 0, 0);
    __builtin_amdgcn_s_wait_asynccnt(0);
#else
    *(v4f*)&tile[r][q * 4] = *(const v4f*)gsrc;
#endif
    __syncthreads();
    int tx = t & 31, ty = t >> 5;   // write coalesced along l
#pragma unroll
    for (int dd = 0; dd < 4; ++dd) {
        int c = ty + 8 * dd;
        int d = dt * 32 + c, l = lt * 32 + tx;
        seqT[((size_t)b * D_ + d) * L_ + l] = (_Float16)tile[tx][c];
    }
}

// ---------------------------------------------------------------------------
// K1: entity embeddings = logsumexp over 4 mention tokens -> f16 [B,NE,D]
// ---------------------------------------------------------------------------
__global__ void entity_emb_kernel(const float* __restrict__ seq,
                                  const int* __restrict__ mpos,
                                  _Float16* __restrict__ eemb16) {
    int e = blockIdx.x, b = blockIdx.y;
    const int* mp = mpos + ((size_t)b * NE_ + e) * M_;
    int p0 = mp[0] + OFFSET_, p1 = mp[1] + OFFSET_;
    int p2 = mp[2] + OFFSET_, p3 = mp[3] + OFFSET_;
    const float* s = seq + (size_t)b * L_ * D_;
    _Float16* out = eemb16 + ((size_t)b * NE_ + e) * D_;
    for (int d = threadIdx.x; d < D_; d += blockDim.x) {
        float v0 = s[(size_t)p0 * D_ + d], v1 = s[(size_t)p1 * D_ + d];
        float v2 = s[(size_t)p2 * D_ + d], v3 = s[(size_t)p3 * D_ + d];
        float mx = fmaxf(fmaxf(v0, v1), fmaxf(v2, v3));
        float sm = expf(v0 - mx) + expf(v1 - mx) + expf(v2 - mx) + expf(v3 - mx);
        out[d] = (_Float16)(mx + logf(sm));
    }
}

// ---------------------------------------------------------------------------
// K2: entity attention = mean over 4 mention rows. f32 [B,NE,H,L]
// ---------------------------------------------------------------------------
__global__ void entity_att_kernel(const float* __restrict__ att,
                                  const int* __restrict__ mpos,
                                  float* __restrict__ eatts) {
    int h = blockIdx.x, e = blockIdx.y, b = blockIdx.z;
    const int* mp = mpos + ((size_t)b * NE_ + e) * M_;
    int p0 = mp[0] + OFFSET_, p1 = mp[1] + OFFSET_;
    int p2 = mp[2] + OFFSET_, p3 = mp[3] + OFFSET_;
    const float* a = att + (((size_t)b * H_ + h) * L_) * L_;
    float* out = eatts + (((size_t)(b * NE_ + e)) * H_ + h) * L_;
    for (int l = threadIdx.x; l < L_; l += blockDim.x)
        out[l] = 0.25f * (a[(size_t)p0 * L_ + l] + a[(size_t)p1 * L_ + l] +
                          a[(size_t)p2 * L_ + l] + a[(size_t)p3 * L_ + l]);
}

// ---------------------------------------------------------------------------
// K3: per-PAIR normalized attention -> f16 [B,P,L] (never build [B,NE,NE,L])
// ---------------------------------------------------------------------------
__global__ void pair_att_kernel(const float* __restrict__ eatts,
                                const int* __restrict__ hts,
                                _Float16* __restrict__ patt16) {
    int p = blockIdx.x, b = blockIdx.y;
    int hidx = hts[((size_t)b * P_ + p) * 2 + 0];
    int tidx = hts[((size_t)b * P_ + p) * 2 + 1];
    const float* Ah = eatts + (((size_t)(b * NE_ + hidx)) * H_) * L_;
    const float* At = eatts + (((size_t)(b * NE_ + tidx)) * H_) * L_;
    __shared__ float q[L_];
    __shared__ float tot;
    if (threadIdx.x == 0) tot = 0.0f;
    __syncthreads();
    float psum = 0.0f;
    for (int l = threadIdx.x; l < L_; l += blockDim.x) {
        float s = 0.0f;
#pragma unroll
        for (int h = 0; h < H_; ++h)
            s += Ah[(size_t)h * L_ + l] * At[(size_t)h * L_ + l];
        s *= (1.0f / H_);
        q[l] = s;
        psum += s;
    }
    atomicAdd(&tot, psum);
    __syncthreads();
    float inv = 1.0f / (tot + 1e-5f);
    _Float16* out = patt16 + ((size_t)b * P_ + p) * L_;
    for (int l = threadIdx.x; l < L_; l += blockDim.x)
        out[l] = (_Float16)(q[l] * inv);
}

// ---------------------------------------------------------------------------
// K4: rs = pairAtt (512x1024) @ seq (1024x768). One wave -> 16x64 output,
// A fragment reused across 4 WMMAs; all operands pre-packed f16.
// ---------------------------------------------------------------------------
__global__ void rs_gemm_kernel(const _Float16* __restrict__ patt16,
                               const _Float16* __restrict__ seqT16,
                               _Float16* __restrict__ rs16) {
    int b = blockIdx.z, m0 = blockIdx.y * 16, n0 = blockIdx.x * 64;
    int lane = threadIdx.x & 31;
    int rlo = lane & 15, hi = (lane >> 4) & 1;
    int hiA = hi * 8, hiB = hi * 16;
    const _Float16* arow = patt16 + ((size_t)b * P_ + m0 + rlo) * L_;
    const _Float16* bcol = seqT16 + ((size_t)b * D_ + n0 + rlo) * L_;
    v8f c0 = {}, c1 = {}, c2 = {}, c3 = {};
    for (int kk = 0; kk < L_; kk += 32) {
        __builtin_prefetch(arow + kk + 256, 0, 1);
        v16h a = load_fragA(arow + kk, hiA);
        v16h w0 = load_fragB(bcol + kk, hiB);
        v16h w1 = load_fragB(bcol + 16 * L_ + kk, hiB);
        v16h w2 = load_fragB(bcol + 32 * L_ + kk, hiB);
        v16h w3 = load_fragB(bcol + 48 * L_ + kk, hiB);
        c0 = __builtin_amdgcn_wmma_f32_16x16x32_f16(false, a, false, w0, (short)0, c0, false, false);
        c1 = __builtin_amdgcn_wmma_f32_16x16x32_f16(false, a, false, w1, (short)0, c1, false, false);
        c2 = __builtin_amdgcn_wmma_f32_16x16x32_f16(false, a, false, w2, (short)0, c2, false, false);
        c3 = __builtin_amdgcn_wmma_f32_16x16x32_f16(false, a, false, w3, (short)0, c3, false, false);
    }
    _Float16* out = rs16 + ((size_t)b * P_) * D_;
#pragma unroll
    for (int r = 0; r < 8; ++r) {
        size_t mr = (size_t)(m0 + r + hi * 8) * D_ + n0 + rlo;
        out[mr]      = (_Float16)c0[r];
        out[mr + 16] = (_Float16)c1[r];
        out[mr + 32] = (_Float16)c2[r];
        out[mr + 48] = (_Float16)c3[r];
    }
}

// ---------------------------------------------------------------------------
// K5/K6: z = tanh(concat(gather(eemb,hts[:,sel]), rs) @ W + b) -> f16 [B*P,D]
// One wave -> 16x64 output. K never straddles the 768 boundary (768 % 32 == 0).
// ---------------------------------------------------------------------------
__global__ void extractor_kernel(const _Float16* __restrict__ eemb16,
                                 const _Float16* __restrict__ rs16,
                                 const int* __restrict__ hts,
                                 const _Float16* __restrict__ WT, // [768][1536]
                                 const float* __restrict__ bias,  // [768] f32
                                 _Float16* __restrict__ z16,      // [B*P][768]
                                 int sel) {
    int m0 = blockIdx.y * 16, n0 = blockIdx.x * 64;
    int lane = threadIdx.x & 31;
    int rlo = lane & 15, hi = (lane >> 4) & 1;
    int hiA = hi * 8, hiB = hi * 16;
    int m = m0 + rlo, bb = m >> 9, pp = m & (P_ - 1);
    int ent = hts[((size_t)bb * P_ + pp) * 2 + sel];
    const _Float16* hrow = eemb16 + ((size_t)bb * NE_ + ent) * D_;
    const _Float16* rrow = rs16 + ((size_t)bb * P_ + pp) * D_;
    const _Float16* wcol = WT + (size_t)(n0 + rlo) * KEX_;
    v8f c0 = {}, c1 = {}, c2 = {}, c3 = {};
    for (int kk = 0; kk < KEX_; kk += 32) {
        const _Float16* ap = (kk < D_) ? (hrow + kk) : (rrow + (kk - D_));
        __builtin_prefetch(wcol + kk + 256, 0, 1);
        v16h a = load_fragA(ap, hiA);
        v16h w0 = load_fragB(wcol + kk, hiB);
        v16h w1 = load_fragB(wcol + (size_t)16 * KEX_ + kk, hiB);
        v16h w2 = load_fragB(wcol + (size_t)32 * KEX_ + kk, hiB);
        v16h w3 = load_fragB(wcol + (size_t)48 * KEX_ + kk, hiB);
        c0 = __builtin_amdgcn_wmma_f32_16x16x32_f16(false, a, false, w0, (short)0, c0, false, false);
        c1 = __builtin_amdgcn_wmma_f32_16x16x32_f16(false, a, false, w1, (short)0, c1, false, false);
        c2 = __builtin_amdgcn_wmma_f32_16x16x32_f16(false, a, false, w2, (short)0, c2, false, false);
        c3 = __builtin_amdgcn_wmma_f32_16x16x32_f16(false, a, false, w3, (short)0, c3, false, false);
    }
#pragma unroll
    for (int r = 0; r < 8; ++r) {
        int mr = m0 + r + hi * 8;
        size_t o = (size_t)mr * D_ + n0 + rlo;
        z16[o]      = (_Float16)tanhf(c0[r] + bias[n0 + rlo]);
        z16[o + 16] = (_Float16)tanhf(c1[r] + bias[n0 + rlo + 16]);
        z16[o + 32] = (_Float16)tanhf(c2[r] + bias[n0 + rlo + 32]);
        z16[o + 48] = (_Float16)tanhf(c3[r] + bias[n0 + rlo + 48]);
    }
}

// ---------------------------------------------------------------------------
// K7: grouped bilinear classifier. For k = g*4096 + i*64 + j:
//   A[m,k] = zs[m,g*64+i] * zo[m,g*64+j]
// built in registers via packed f16 mul (v_pk_mul_f16); each 8-element half
// of the fragment has constant (g,i) since k-offsets are multiples of 8.
// One wave computes all 7 n-tiles (16x112), reusing A 7x.
// ---------------------------------------------------------------------------
__global__ void bilinear_kernel(const _Float16* __restrict__ zs16,
                                const _Float16* __restrict__ zo16,
                                const _Float16* __restrict__ WbT, // [112][49152]
                                const float* __restrict__ bbi,    // [97]
                                float* __restrict__ out) {        // [B*P][97]
    int m0 = blockIdx.x * 16;
    int lane = threadIdx.x & 31;
    int rlo = lane & 15, hi = (lane >> 4) & 1;
    int hiA = hi * 8, hiB = hi * 16;
    const _Float16* zsr = zs16 + (size_t)(m0 + rlo) * D_;
    const _Float16* zor = zo16 + (size_t)(m0 + rlo) * D_;
    const _Float16* wb = WbT + (size_t)rlo * KBI_;
    v8f c[7] = {};
    for (int kk = 0; kk < KBI_; kk += 32) {
        int ka0 = kk + hiA, ka1 = kk + 16 + hiA;
        int g0 = ka0 >> 12, i0 = (ka0 >> 6) & 63, j0 = ka0 & 63;
        int g1 = ka1 >> 12, i1 = (ka1 >> 6) & 63, j1 = ka1 & 63;
        _Float16 s0 = zsr[g0 * BLK_ + i0];
        _Float16 s1 = zsr[g1 * BLK_ + i1];
        v8h o0 = *(const v8h*)(zor + g0 * BLK_ + j0);
        v8h o1 = *(const v8h*)(zor + g1 * BLK_ + j1);
        v16h a = frag_cat(o0 * s0, o1 * s1);
#pragma unroll
        for (int t = 0; t < 7; ++t) {
            v16h w = load_fragB(wb + (size_t)(t * 16) * KBI_ + kk, hiB);
            c[t] = __builtin_amdgcn_wmma_f32_16x16x32_f16(false, a, false, w,
                                                          (short)0, c[t], false, false);
        }
    }
#pragma unroll
    for (int t = 0; t < 7; ++t) {
        int n = t * 16 + rlo;
        if (n < NL_) {
#pragma unroll
            for (int r = 0; r < 8; ++r) {
                int mr = m0 + r + hi * 8;
                out[(size_t)mr * NL_ + n] = c[t][r] + bbi[n];
            }
        }
    }
}

// ---------------------------------------------------------------------------
extern "C" void kernel_launch(void* const* d_in, const int* in_sizes, int n_in,
                              void* d_out, int out_size, void* d_ws, size_t ws_size,
                              hipStream_t stream) {
    (void)in_sizes; (void)n_in; (void)out_size; (void)ws_size;
    const float* seq  = (const float*)d_in[0]; // [B,L,D]
    const float* att  = (const float*)d_in[1]; // [B,H,L,L]
    const int*   mpos = (const int*)d_in[2];   // [B,NE,M]
    const int*   hts  = (const int*)d_in[3];   // [B,P,2]
    const float* Wh   = (const float*)d_in[4]; // [2D,D]
    const float* bh   = (const float*)d_in[5]; // [D]
    const float* Wt   = (const float*)d_in[6]; // [2D,D]
    const float* bt   = (const float*)d_in[7]; // [D]
    const float* Wbi  = (const float*)d_in[8]; // [D*BLK,NL]
    const float* bbi  = (const float*)d_in[9]; // [NL]
    float* out = (float*)d_out;                // [B,P,NL] f32

    // Workspace carve-up (256B-aligned chunks), ~44 MB total.
    char* w = (char*)d_ws;
    auto alloc = [&](size_t bytes) -> char* {
        char* p = w;
        w += (bytes + 255) & ~(size_t)255;
        return p;
    };
    float*    eatts  = (float*)   alloc((size_t)B_ * NE_ * H_ * L_ * 4); // 8.3MB
    _Float16* patt16 = (_Float16*)alloc((size_t)B_ * P_ * L_ * 2);       // 4.2MB
    _Float16* eemb16 = (_Float16*)alloc((size_t)B_ * NE_ * D_ * 2);
    _Float16* rs16   = (_Float16*)alloc((size_t)B_ * P_ * D_ * 2);       // 3.1MB
    _Float16* zs16   = (_Float16*)alloc((size_t)B_ * P_ * D_ * 2);
    _Float16* zo16   = (_Float16*)alloc((size_t)B_ * P_ * D_ * 2);
    _Float16* seqT16 = (_Float16*)alloc((size_t)B_ * D_ * L_ * 2);       // 6.3MB
    _Float16* whT16  = (_Float16*)alloc((size_t)D_ * KEX_ * 2);          // 2.3MB
    _Float16* wtT16  = (_Float16*)alloc((size_t)D_ * KEX_ * 2);
    _Float16* wbiT16 = (_Float16*)alloc((size_t)NPAD_ * KBI_ * 2);       // 10.5MB

    // Operand packing (one-shot streaming passes)
    weight_transpose_kernel<<<dim3(6, D_), 256, 0, stream>>>(Wh, whT16, KEX_, D_);
    weight_transpose_kernel<<<dim3(6, D_), 256, 0, stream>>>(Wt, wtT16, KEX_, D_);
    weight_transpose_kernel<<<dim3(192, NPAD_), 256, 0, stream>>>(Wbi, wbiT16, KBI_, NL_);
    seq_transpose_kernel<<<dim3(L_ / 32, D_ / 32, B_), 256, 0, stream>>>(seq, seqT16);

    // Gathers / pooling
    entity_emb_kernel<<<dim3(NE_, B_), 256, 0, stream>>>(seq, mpos, eemb16);
    entity_att_kernel<<<dim3(H_, NE_, B_), 256, 0, stream>>>(att, mpos, eatts);
    pair_att_kernel<<<dim3(P_, B_), 256, 0, stream>>>(eatts, hts, patt16);

    // WMMA GEMMs
    rs_gemm_kernel<<<dim3(D_ / 64, P_ / 16, B_), 32, 0, stream>>>(patt16, seqT16, rs16);
    extractor_kernel<<<dim3(D_ / 64, (B_ * P_) / 16), 32, 0, stream>>>(
        eemb16, rs16, hts, whT16, bh, zs16, 0);
    extractor_kernel<<<dim3(D_ / 64, (B_ * P_) / 16), 32, 0, stream>>>(
        eemb16, rs16, hts, wtT16, bt, zo16, 1);
    bilinear_kernel<<<dim3((B_ * P_) / 16), 32, 0, stream>>>(zs16, zo16, wbiT16, bbi, out);
}
